// LadderSABlock_88553635709739
// MI455X (gfx1250) — matrix operand order repaired and checked by reading, not compiled, not run on hardware
//
#include <hip/hip_runtime.h>

// ---------------------------------------------------------------------------
// Swin window-attention block for MI455X (gfx1250, wave32, WMMA)
// One workgroup (256 threads = 8 wave32) handles one 7x7 window (N=49, C=192,
// 6 heads x 32). GEMMs use v_wmma_f32_16x16x32_f16 with f32 accumulation.
// Each wave owns a fixed 16-row tile; A-operands are hoisted into registers.
// Wave id is readfirstlane-broadcast so all tile control flow is scalar.
// ---------------------------------------------------------------------------

typedef __attribute__((ext_vector_type(16))) _Float16 v16h;
typedef __attribute__((ext_vector_type(8)))  float    v8f;

#define DIMC      192
#define NHEADS    6
#define HD        32
#define WS        7
#define NTOK      49
#define NPAD      64
#define HRES      56
#define NWIN_SIDE 8            // 56/7
#define SCALE_Q   0.17677669529663689f   // 32^-0.5

// LDS strides (in _Float16 elements / floats)
#define XN_STRIDE 200   // 192 + 8 pad
#define QK_STRIDE 36    // 32 + 4 pad
#define VT_STRIDE 72    // 64 + 8 pad
#define P_STRIDE  72
#define S_STRIDE  66    // 64 + 2 pad: hi-lane D rows land +16 banks -> conflict-free

// LDS byte offsets (carved from dynamic shared memory)
#define OFF_XA   0                        // 64*200 halves = 25600 B (51200 B reserved) (xn, later attn_out)
#define OFF_Q    51200                    // 6*64*36 halves = 27648 B
#define OFF_K    78848                    // 27648 B
#define OFF_VT   106496                   // 6*32*72 halves = 27648 B
#define OFF_S    134144                   // 64*66 floats  = 16896 B
#define OFF_P    151040                   // 64*72 halves  = 9216 B
#define OFF_TOK  160256                   // 49 ints (token base offsets), pad to 256 B
#define SMEM_BYTES 160512                 // <= 163840 -> 2 workgroups per WGP

// workspace byte offsets
#define WS_QKVW16 0                       // 576*192 f16 = 221184 B
#define WS_PW16   221184                  // 192*192 f16 = 73728 B
#define WS_BIASX  294912                  // 6*49*49 f32 = 57624 B

union U16H { v16h v; uint32_t u[8]; };

// A-operand (16x32 f16): storage row-major [row][k], per ISA A layout:
// lane row = lane%16 ; halves 0..7 -> K=koff..koff+7 ; halves 8..15 -> K=koff+16..+23
__device__ __forceinline__ v16h load_tileA(const _Float16* tileBase, int strideH) {
  const int lane = threadIdx.x & 31;
  const int row  = lane & 15;
  const int koff = (lane >> 4) << 3;           // 0 or 8
  const uint32_t* p0 = (const uint32_t*)(tileBase + row * strideH + koff);
  const uint32_t* p1 = (const uint32_t*)(tileBase + row * strideH + koff + 16);
  U16H a;
#pragma unroll
  for (int i = 0; i < 4; ++i) { a.u[i] = p0[i]; a.u[4 + i] = p1[i]; }
  return a.v;
}

// B-operand (32x16 f16): storage row-major [n][k] (n = output col), per-lane
// n = lane%16, halves 0..7 -> K=koff..koff+7 ; halves 8..15 -> K=koff+16..+23 (koff = 8*(lane>=16))
__device__ __forceinline__ v16h load_tileB(const _Float16* tileBase, int strideH) {
  const int lane  = threadIdx.x & 31;
  const int n     = lane & 15;
  const int koff  = (lane >> 4) << 3;          // 0 or 8
  const uint32_t* p = (const uint32_t*)(tileBase + n * strideH + koff);
  U16H b;
#pragma unroll
  for (int i = 0; i < 4; ++i) { b.u[i] = p[i]; b.u[4 + i] = p[8 + i]; }
  return b.v;
}

__device__ __forceinline__ v8f wmma_f16(v16h a, v16h b, v8f c) {
  return __builtin_amdgcn_wmma_f32_16x16x32_f16(false, a, false, b, (short)0, c,
                                                false, false);
}

// ---------------------------------------------------------------------------
// Pre-pass: convert weights to f16, expand relative-position bias table
// ---------------------------------------------------------------------------
__global__ void swin_prepass(const float* __restrict__ qkv_w,
                             const float* __restrict__ proj_w,
                             const float* __restrict__ rpb,
                             _Float16* __restrict__ qkvw16,
                             _Float16* __restrict__ pw16,
                             float* __restrict__ biasx) {
  const int i = blockIdx.x * 256 + threadIdx.x;
  const int NQ = 3 * DIMC * DIMC;            // 110592
  const int NP = DIMC * DIMC;                // 36864
  const int NB = NHEADS * NTOK * NTOK;       // 14406
  if (i < NQ) {
    qkvw16[i] = (_Float16)qkv_w[i];
  } else if (i < NQ + NP) {
    pw16[i - NQ] = (_Float16)proj_w[i - NQ];
  } else if (i < NQ + NP + NB) {
    int j = i - NQ - NP;
    int h  = j / (NTOK * NTOK);
    int nm = j % (NTOK * NTOK);
    int n = nm / NTOK, m = nm % NTOK;
    int i1 = n / WS, j1 = n % WS, i2 = m / WS, j2 = m % WS;
    int idx = (i1 - i2 + WS - 1) * (2 * WS - 1) + (j1 - j2 + WS - 1);
    biasx[j] = rpb[idx * NHEADS + h];
  }
}

// ---------------------------------------------------------------------------
// Main kernel: one workgroup per window
// ---------------------------------------------------------------------------
__global__ __launch_bounds__(256, 2)
void swin_attn(const float* __restrict__ x,
               const float* __restrict__ gamma,
               const float* __restrict__ beta,
               const float* __restrict__ qkv_b,
               const float* __restrict__ proj_b,
               const _Float16* __restrict__ qkvw16,
               const _Float16* __restrict__ pw16,
               const float* __restrict__ biasx,
               float* __restrict__ out) {
  extern __shared__ char smem[];
  _Float16* sXA  = (_Float16*)(smem + OFF_XA);   // xn, reused as attn_out
  _Float16* sQ   = (_Float16*)(smem + OFF_Q);
  _Float16* sK   = (_Float16*)(smem + OFF_K);
  _Float16* sVT  = (_Float16*)(smem + OFF_VT);
  float*    sS   = (float*)   (smem + OFF_S);
  _Float16* sP   = (_Float16*)(smem + OFF_P);
  int*      sTok = (int*)     (smem + OFF_TOK);

  const int tid  = threadIdx.x;
  const int lane = tid & 31;
  // wave id, broadcast to SGPR so all tile control flow is provably uniform
  const int wv   = __builtin_amdgcn_readfirstlane(tid >> 5);    // 0..7
  const int ln15 = lane & 15;
  const int hi8  = (lane >> 4) << 3;            // 0 or 8 (D-row offset)

  const int wid = blockIdx.x;
  const int b   = wid >> 6;                     // image
  const int ww  = wid & 63;
  const int wy  = ww >> 3, wx = ww & 7;

  // ---- Stage 1: load + LayerNorm -> f16 xn in LDS (rows 49..63 zero) ----
  for (int r = wv; r < NPAD; r += 8) {
    if (r < NTOK) {
      const int trow = wy * WS + r / WS;
      const int tcol = wx * WS + r % WS;
      const int base = (b * (HRES * HRES) + trow * HRES + tcol) * DIMC;
      if (lane == 0) sTok[r] = base;            // token base offset table
      const float* xp = x + base;
      float v[6];
      float s = 0.f, s2 = 0.f;
#pragma unroll
      for (int j = 0; j < 6; ++j) {
        v[j] = xp[lane + 32 * j];
        s += v[j]; s2 += v[j] * v[j];
      }
#pragma unroll
      for (int m = 16; m >= 1; m >>= 1) {
        s  += __shfl_xor(s,  m, 32);
        s2 += __shfl_xor(s2, m, 32);
      }
      const float mean = s * (1.f / DIMC);
      const float var  = s2 * (1.f / DIMC) - mean * mean;
      const float inv  = rsqrtf(var + 1e-5f);
#pragma unroll
      for (int j = 0; j < 6; ++j) {
        const int c = lane + 32 * j;
        sXA[r * XN_STRIDE + c] =
            (_Float16)((v[j] - mean) * inv * gamma[c] + beta[c]);
      }
    } else {
#pragma unroll
      for (int j = 0; j < 6; ++j) sXA[r * XN_STRIDE + lane + 32 * j] = (_Float16)0.f;
    }
  }
  __syncthreads();

  // ---- Stage 2: QKV GEMM  (64x192) @ (192x576) ----
  // Wave layout: rt = wv&3 (fixed row tile, A hoisted), column half = wv>>2.
  {
    const int rt  = wv & 3;
    const int cth = wv >> 2;
    v16h aq[6];
#pragma unroll
    for (int kk = 0; kk < 6; ++kk)
      aq[kk] = load_tileA(&sXA[rt * 16 * XN_STRIDE + kk * 32], XN_STRIDE);

    const int rowB = rt * 16 + hi8;
    for (int ct = cth * 18; ct < cth * 18 + 18; ++ct) {
      v8f acc = {};
#pragma unroll
      for (int kk = 0; kk < 6; ++kk) {
        v16h bm = load_tileB(&qkvw16[(size_t)(ct * 16) * DIMC + kk * 32], DIMC);
        acc = wmma_f16(aq[kk], bm, acc);
      }
      // scalar (wave-uniform) decomposition of the output column block
      const int s  = ct / 12;              // 0=q 1=k 2=v
      const int hh = (ct % 12) >> 1;       // head
      const int d  = ((ct & 1) << 4) + ln15;
      const float bv = qkv_b[ct * 16 + ln15];
      if (s == 0) {
#pragma unroll
        for (int r = 0; r < 8; ++r)
          sQ[(hh * NPAD + rowB + r) * QK_STRIDE + d] =
              (_Float16)((acc[r] + bv) * SCALE_Q);
      } else if (s == 1) {
#pragma unroll
        for (int r = 0; r < 8; ++r)
          sK[(hh * NPAD + rowB + r) * QK_STRIDE + d] = (_Float16)(acc[r] + bv);
      } else {
#pragma unroll
        for (int r = 0; r < 8; ++r)
          sVT[(hh * HD + d) * VT_STRIDE + rowB + r] = (_Float16)(acc[r] + bv);
      }
    }
  }
  __syncthreads();

  // ---- Stage 3: attention per head ----
  for (int h = 0; h < NHEADS; ++h) {
    // S = (Q*scale) @ K^T : 4x4 output tiles, K-dim = 32 (one WMMA each)
    {
      const int rt  = wv >> 1;
      const int cth = wv & 1;               // 2 column tiles per wave
      const int rowB = rt * 16 + hi8;
      v16h a = load_tileA(&sQ[(h * NPAD + rt * 16) * QK_STRIDE], QK_STRIDE);
#pragma unroll
      for (int cc = 0; cc < 2; ++cc) {
        const int ctm = cth * 2 + cc;
        v16h bm = load_tileB(&sK[(h * NPAD + ctm * 16) * QK_STRIDE], QK_STRIDE);
        v8f acc = {};
        acc = wmma_f16(a, bm, acc);
        const int m = ctm * 16 + ln15;
#pragma unroll
        for (int r = 0; r < 8; ++r) sS[(rowB + r) * S_STRIDE + m] = acc[r];
      }
    }
    __syncthreads();

    // softmax rows (one wave per row, 2 chunks of 32 columns)
    for (int r = wv; r < NPAD; r += 8) {
      if (r < NTOK) {
        const float* bxr = &biasx[(h * NTOK + r) * NTOK];
        const int m1 = lane + 32;
        float v0 = (lane < NTOK) ? sS[r * S_STRIDE + lane] + bxr[lane] : -1e30f;
        float v1 = (m1 < NTOK)   ? sS[r * S_STRIDE + m1]   + bxr[m1]   : -1e30f;
        float mx = fmaxf(v0, v1);
#pragma unroll
        for (int m = 16; m >= 1; m >>= 1) mx = fmaxf(mx, __shfl_xor(mx, m, 32));
        float e0 = (lane < NTOK) ? __expf(v0 - mx) : 0.f;
        float e1 = (m1 < NTOK)   ? __expf(v1 - mx) : 0.f;
        float sm = e0 + e1;
#pragma unroll
        for (int m = 16; m >= 1; m >>= 1) sm += __shfl_xor(sm, m, 32);
        const float inv = 1.f / sm;
        sP[r * P_STRIDE + lane] = (_Float16)(e0 * inv);
        sP[r * P_STRIDE + m1]   = (_Float16)(e1 * inv);
      } else {
        sP[r * P_STRIDE + lane]      = (_Float16)0.f;
        sP[r * P_STRIDE + lane + 32] = (_Float16)0.f;
      }
    }
    __syncthreads();

    // O = P @ V : 4x2 output tiles (one per wave), K-dim = 64 (2 WMMA)
    {
      const int rt = wv >> 1, ctd = wv & 1;
      v8f acc = {};
#pragma unroll
      for (int kk = 0; kk < 2; ++kk) {
        v16h a  = load_tileA(&sP[rt * 16 * P_STRIDE + kk * 32], P_STRIDE);
        v16h bm = load_tileB(&sVT[(h * HD + ctd * 16) * VT_STRIDE + kk * 32], VT_STRIDE);
        acc = wmma_f16(a, bm, acc);
      }
      const int d    = ctd * 16 + ln15;
      const int rowB = rt * 16 + hi8;
#pragma unroll
      for (int r = 0; r < 8; ++r)
        sXA[(rowB + r) * XN_STRIDE + h * HD + d] = (_Float16)acc[r];
    }
    __syncthreads();
  }

  // ---- Stage 4: proj GEMM + bias + residual + store ----
  {
    const int rt  = wv & 3;
    const int cth = wv >> 2;
    v16h ap[6];
#pragma unroll
    for (int kk = 0; kk < 6; ++kk)
      ap[kk] = load_tileA(&sXA[rt * 16 * XN_STRIDE + kk * 32], XN_STRIDE);

    const int rowB = rt * 16 + hi8;
    for (int ct = cth * 6; ct < cth * 6 + 6; ++ct) {
      v8f acc = {};
#pragma unroll
      for (int kk = 0; kk < 6; ++kk) {
        v16h bm = load_tileB(&pw16[(size_t)(ct * 16) * DIMC + kk * 32], DIMC);
        acc = wmma_f16(ap[kk], bm, acc);
      }
      const int o    = ct * 16 + ln15;
      const float pb = proj_b[o];
      if (rt < 3) {
        // rows rt*16 .. rt*16+15 are all < 49: unconditional stores
#pragma unroll
        for (int r = 0; r < 8; ++r) {
          const size_t gi = (size_t)sTok[rowB + r] + o;
          out[gi] = acc[r] + pb + x[gi];
        }
      } else {
        // rt == 3: only row 48 (lo lanes, r == 0) is in-bounds
        if (hi8 == 0) {
          const size_t gi = (size_t)sTok[48] + o;
          out[gi] = acc[0] + pb + x[gi];
        }
      }
    }
  }
}

// ---------------------------------------------------------------------------
extern "C" void kernel_launch(void* const* d_in, const int* in_sizes, int n_in,
                              void* d_out, int out_size, void* d_ws, size_t ws_size,
                              hipStream_t stream) {
  const float* x      = (const float*)d_in[0];
  const float* gamma  = (const float*)d_in[1];
  const float* beta   = (const float*)d_in[2];
  const float* qkv_w  = (const float*)d_in[3];
  const float* qkv_b  = (const float*)d_in[4];
  const float* proj_w = (const float*)d_in[5];
  const float* proj_b = (const float*)d_in[6];
  const float* rpb    = (const float*)d_in[7];
  float* out          = (float*)d_out;

  _Float16* qkvw16 = (_Float16*)((char*)d_ws + WS_QKVW16);
  _Float16* pw16   = (_Float16*)((char*)d_ws + WS_PW16);
  float*    biasx  = (float*)   ((char*)d_ws + WS_BIASX);

  const int preTotal = 3 * DIMC * DIMC + DIMC * DIMC + NHEADS * NTOK * NTOK;
  swin_prepass<<<(preTotal + 255) / 256, 256, 0, stream>>>(
      qkv_w, proj_w, rpb, qkvw16, pw16, biasx);

  const int numWindows = 64 * NWIN_SIDE * NWIN_SIDE;  // 4096
  swin_attn<<<numWindows, 256, SMEM_BYTES, stream>>>(
      x, gamma, beta, qkv_b, proj_b, qkvw16, pw16, biasx, out);
}
